// Beit_9560597201107
// MI455X (gfx1250) — matrix operand order repaired and checked by reading, not compiled, not run on hardware
//
#include <hip/hip_runtime.h>
#include <math.h>

// ---------------------------------------------------------------------------
// BEiT attention block for MI455X (gfx1250), bf16 WMMA with fp32 accumulate.
// Shapes: B=64, N=197, DIM=768, HEADS=12, HEAD_DIM=64, NBS=10, M=B*N=12608.
// ---------------------------------------------------------------------------

typedef __bf16 bf16_t;
typedef __attribute__((ext_vector_type(16))) __bf16 v16bf;
typedef __attribute__((ext_vector_type(8)))  float  v8f;

#define BEIT_B      64
#define BEIT_N      197
#define BEIT_DIM    768
#define BEIT_HEADS  12
#define BEIT_HD     64
#define BEIT_M      (BEIT_B * BEIT_N)      // 12608
#define BEIT_3DIM   (3 * BEIT_DIM)        // 2304
#define NPAD        208                    // 13 tiles of 16
#define NTILES      13

__device__ inline v8f vzero8() {
    v8f z = {0.f, 0.f, 0.f, 0.f, 0.f, 0.f, 0.f, 0.f};
    return z;
}

__device__ inline v8f wmma_bf16(v16bf a, v16bf b, v8f c) {
    // D = A(16x32 bf16) * B(32x16 bf16) + C(16x16 f32)
    return __builtin_amdgcn_wmma_f32_16x16x32_bf16(
        /*neg_a=*/false, a, /*neg_b=*/false, b,
        /*c_mod=*/(short)0, c, /*reuse_a=*/false, /*reuse_b=*/false);
}

// A-fragment swizzle: element (m, k) of a 16x32 tile lives at
//   lane = m + 16*((k>>3)&1),  e = ((k>>4)<<3) + (k&7)
// B-fragment swizzle: element (k, n) of a 32x16 tile lives at
//   lane = n + 16*((k>>3)&1), e = ((k>>4)<<3) + (k&7)  (attn: k = 16*(lane>>4) + e, A and B)
// Each lane then reads its 16 bf16 (32 B, contiguous) fragment from LDS.

// ---------------------------------------------------------------------------
// Elementwise f32 -> bf16 cast
// ---------------------------------------------------------------------------
__global__ __launch_bounds__(256) void cast_f32_bf16(
    const float* __restrict__ src, bf16_t* __restrict__ dst, int n) {
    int i = blockIdx.x * 256 + threadIdx.x;
    if (i < n) dst[i] = (bf16_t)src[i];
}

// ---------------------------------------------------------------------------
// rel_bias[h][n][m] = rel_table[rel_index[n][m]][h]   (fp32, [12][197][197])
// ---------------------------------------------------------------------------
__global__ __launch_bounds__(256) void relbias_build(
    const float* __restrict__ table, const int* __restrict__ ridx,
    float* __restrict__ relb) {
    int i = blockIdx.x * 256 + threadIdx.x;     // n*197 + m
    if (i < BEIT_N * BEIT_N) {
        int idx = ridx[i];
        #pragma unroll
        for (int h = 0; h < BEIT_HEADS; ++h)
            relb[(size_t)h * (BEIT_N * BEIT_N) + i] = table[idx * BEIT_HEADS + h];
    }
}

// ---------------------------------------------------------------------------
// QKV GEMM: [12608 x 768] x [768 x 2304]^T, bf16 in, fp32 acc.
// Epilogue: +q_bias (then *0.125), +v_bias, split into q/k/v [B,H,N,64] bf16.
// Block tile 128x128, 8 waves of 32x64, K-step 32.
// ---------------------------------------------------------------------------
__global__ __launch_bounds__(256) void qkv_gemm(
    const bf16_t* __restrict__ xb, const bf16_t* __restrict__ wq,
    const float* __restrict__ q_bias, const float* __restrict__ v_bias,
    const int* __restrict__ b_idx,
    bf16_t* __restrict__ qo, bf16_t* __restrict__ ko, bf16_t* __restrict__ vo) {
    __shared__ alignas(32) bf16_t alds[8 * 512];   // 128 x 32, A-swizzled
    __shared__ alignas(32) bf16_t blds[8 * 512];   // 32 x 128, B-swizzled

    const int tid  = threadIdx.x;
    const int wv   = tid >> 5, lane = tid & 31;
    const int col0 = blockIdx.x * 128;             // 0..2176
    const int row0 = blockIdx.y * 128;             // 0..12544
    const int wm = wv >> 1, wn = wv & 1;

    v8f acc[2][4];
    #pragma unroll
    for (int i = 0; i < 2; ++i)
        #pragma unroll
        for (int j = 0; j < 4; ++j) acc[i][j] = vzero8();

    for (int k0 = 0; k0 < BEIT_DIM; k0 += 32) {
        // ---- stage A tile (x rows), 2048 dwords, coalesced pairs ----
        #pragma unroll
        for (int i = 0; i < 8; ++i) {
            int u  = tid + (i << 8);
            int ml = u >> 4;                // 0..127
            int k  = (u & 15) << 1;         // even 0..30
            int row = row0 + ml; if (row > BEIT_M - 1) row = BEIT_M - 1;
            unsigned val = *(const unsigned*)(xb + (size_t)row * BEIT_DIM + k0 + k);
            int l = (ml & 15) + (((k >> 3) & 1) << 4);
            int e = ((k >> 4) << 3) + (k & 7);
            *(unsigned*)(alds + (ml >> 4) * 512 + l * 16 + e) = val;
        }
        // ---- stage B tile (weight rows), 2048 dwords ----
        #pragma unroll
        for (int i = 0; i < 8; ++i) {
            int u  = tid + (i << 8);
            int nl = u >> 4;                // 0..127
            int k  = (u & 15) << 1;
            unsigned val = *(const unsigned*)(wq + (size_t)(col0 + nl) * BEIT_DIM + k0 + k);
            int l = (nl & 15) + (((k >> 3) & 1) << 4);
            int e = ((k >> 4) << 3) + (k & 7);
            *(unsigned*)(blds + (nl >> 4) * 512 + l * 16 + e) = val;
        }
        __syncthreads();

        v16bf af[2], bf[4];
        #pragma unroll
        for (int i = 0; i < 2; ++i)
            af[i] = *(const v16bf*)(alds + (wm * 2 + i) * 512 + lane * 16);
        #pragma unroll
        for (int j = 0; j < 4; ++j)
            bf[j] = *(const v16bf*)(blds + (wn * 4 + j) * 512 + lane * 16);
        #pragma unroll
        for (int i = 0; i < 2; ++i)
            #pragma unroll
            for (int j = 0; j < 4; ++j)
                acc[i][j] = wmma_bf16(af[i], bf[j], acc[i][j]);
        __syncthreads();
    }

    // ---- epilogue: bias, scale, q/k/v split to [B,H,N,64] bf16 ----
    const int nlane = lane & 15;
    const int rh    = (lane >> 4) << 3;
    const int sec   = col0 / BEIT_DIM;     // 0=q, 1=k, 2=v (tile never straddles)
    #pragma unroll
    for (int i = 0; i < 2; ++i) {
        #pragma unroll
        for (int j = 0; j < 4; ++j) {
            int colg = col0 + wn * 64 + j * 16 + nlane;
            int cl   = colg - sec * BEIT_DIM;          // 0..767
            int h = cl >> 6, d = cl & 63;
            #pragma unroll
            for (int r = 0; r < 8; ++r) {
                int row = row0 + wm * 32 + i * 16 + rh + r;
                if (row >= BEIT_M) continue;
                int bsm = row / BEIT_N;
                int tok = row - bsm * BEIT_N;
                int bi  = b_idx[bsm];
                float val = acc[i][j][r];
                size_t oidx = ((size_t)(bsm * BEIT_HEADS + h) * BEIT_N + tok) * BEIT_HD + d;
                if (sec == 0) {
                    val = (val + q_bias[bi * BEIT_DIM + cl]) * 0.125f;  // HEAD_DIM^-0.5
                    qo[oidx] = (bf16_t)val;
                } else if (sec == 1) {
                    ko[oidx] = (bf16_t)val;
                } else {
                    val = val + v_bias[bi * BEIT_DIM + cl];
                    vo[oidx] = (bf16_t)val;
                }
            }
        }
    }
}

// ---------------------------------------------------------------------------
// Attention: one block per (b,h). S = q k^T (+rel bias) -> softmax -> P v.
// K/V staged pre-swizzled in LDS; each wave owns 16 query rows at a time.
// ---------------------------------------------------------------------------
__global__ __launch_bounds__(256) void attn_kernel(
    const bf16_t* __restrict__ qb, const bf16_t* __restrict__ kb,
    const bf16_t* __restrict__ vb, const float* __restrict__ relb,
    bf16_t* __restrict__ ao) {
    __shared__ alignas(32) bf16_t kls[26 * 512];   // K^T B-frags: (ntile,kc) x 512
    __shared__ alignas(32) bf16_t vls[28 * 512];   // V   B-frags: (kc,dt)  x 512
    __shared__ alignas(32) bf16_t pls[8 * 512];    // per-wave P reswizzle buffer

    const int tid = threadIdx.x, wv = tid >> 5, lane = tid & 31;
    const int bh = blockIdx.x;                     // b*12 + h
    const int b  = bh / BEIT_HEADS, h = bh - b * BEIT_HEADS;
    const size_t base = (size_t)bh * BEIT_N * BEIT_HD;

    // ---- stage K as B-matrix for S: B[d][m_key], d is the K-dim (64) ----
    #pragma unroll 2
    for (int i = 0; i < 26; ++i) {
        int u  = tid + (i << 8);        // 0..6655
        int d  = (u & 31) << 1;         // even 0..62
        int nk = u >> 5;                // key row 0..207
        unsigned val = 0;
        if (nk < BEIT_N) val = *(const unsigned*)(kb + base + (size_t)nk * BEIT_HD + d);
        int kc = d >> 5, ki = d & 31;
        int l = (nk & 15) + ((ki >> 4) << 4);
        int e = ki & 15;
        *(unsigned*)(kls + ((nk >> 4) * 2 + kc) * 512 + l * 16 + e) = val;
    }
    // ---- stage V as B-matrix for O: B[n_key][d], n_key is the K-dim (208) ----
    #pragma unroll 2
    for (int i = 0; i < 26; ++i) {
        int u  = tid + (i << 8);
        int d  = (u & 31) << 1;
        int kk = u >> 5;                // key row 0..207
        unsigned val = 0;
        if (kk < BEIT_N) val = *(const unsigned*)(vb + base + (size_t)kk * BEIT_HD + d);
        bf16_t lo = ((const bf16_t*)&val)[0];
        bf16_t hi = ((const bf16_t*)&val)[1];
        int kc = kk >> 5, ki = kk & 31;
        int st = kc * 4 + (d >> 4);
        int l  = (d & 15) + ((ki >> 4) << 4);
        int e  = ki & 15;
        vls[st * 512 + l * 16 + e]       = lo;
        vls[st * 512 + (l + 1) * 16 + e] = hi;
    }
    __syncthreads();

    const int nlane = lane & 15;
    const int rh    = (lane >> 4) << 3;

    for (int mt = wv; mt < NTILES; mt += 8) {
        // q A-fragments straight from global (contiguous bf16 pairs per elem)
        int qrow = mt * 16 + nlane; if (qrow > BEIT_N - 1) qrow = BEIT_N - 1;
        v16bf aq[2];
        #pragma unroll
        for (int dk = 0; dk < 2; ++dk) {
            union { v16bf v; unsigned u[8]; } t;
            #pragma unroll
            for (int jj = 0; jj < 8; ++jj) {
                int k = (jj << 1) + ((lane >> 4) << 4);
                t.u[jj] = *(const unsigned*)(qb + base + (size_t)qrow * BEIT_HD + dk * 32 + k);
            }
            aq[dk] = t.v;
        }

        // ---- S = q k^T + rel_bias, masked ----
        v8f sreg[NTILES];
        for (int nt = 0; nt < NTILES; ++nt) {
            v8f s = vzero8();
            #pragma unroll
            for (int dk = 0; dk < 2; ++dk) {
                v16bf kf = *(const v16bf*)(kls + (nt * 2 + dk) * 512 + lane * 16);
                s = wmma_bf16(aq[dk], kf, s);
            }
            int col = nt * 16 + nlane;
            #pragma unroll
            for (int r = 0; r < 8; ++r) {
                int rowm = mt * 16 + rh + r;
                if (rowm < BEIT_N && col < BEIT_N)
                    s[r] += relb[((size_t)h * BEIT_N + rowm) * BEIT_N + col];
                else
                    s[r] = -__builtin_inff();
            }
            sreg[nt] = s;
        }

        // ---- row max (within 16-lane groups) ----
        v8f mx = sreg[0];
        for (int nt = 1; nt < NTILES; ++nt)
            #pragma unroll
            for (int r = 0; r < 8; ++r) mx[r] = fmaxf(mx[r], sreg[nt][r]);
        #pragma unroll
        for (int r = 0; r < 8; ++r)
            for (int off = 1; off < 16; off <<= 1)
                mx[r] = fmaxf(mx[r], __shfl_xor(mx[r], off, 32));

        // ---- streamed P·V: reswizzle 16x32 P chunks through LDS ----
        v8f rsum = vzero8();
        v8f ov[4];
        #pragma unroll
        for (int dt = 0; dt < 4; ++dt) ov[dt] = vzero8();

        for (int c = 0; c < 7; ++c) {
            #pragma unroll
            for (int sub = 0; sub < 2; ++sub) {
                int nt = c * 2 + sub;
                v8f p = vzero8();
                if (nt < NTILES) {
                    #pragma unroll
                    for (int r = 0; r < 8; ++r) p[r] = __expf(sreg[nt][r] - mx[r]);
                }
                #pragma unroll
                for (int r = 0; r < 8; ++r) {
                    rsum[r] += p[r];
                    int kl = sub * 16 + nlane;                     // K-pos in chunk
                    int la = (rh + r) + ((kl >> 4) << 4);          // A-lane for row
                    int e  = kl & 15;
                    pls[wv * 512 + la * 16 + e] = (bf16_t)p[r];
                }
            }
            asm volatile("s_wait_dscnt 0" ::: "memory");  // same-wave LDS RAW
            v16bf pa = *(const v16bf*)(pls + wv * 512 + lane * 16);
            #pragma unroll
            for (int dt = 0; dt < 4; ++dt) {
                v16bf vf = *(const v16bf*)(vls + (c * 4 + dt) * 512 + lane * 16);
                ov[dt] = wmma_bf16(pa, vf, ov[dt]);
            }
        }

        // ---- normalize and store [B,N,768] bf16 ----
        #pragma unroll
        for (int r = 0; r < 8; ++r)
            for (int off = 1; off < 16; off <<= 1)
                rsum[r] += __shfl_xor(rsum[r], off, 32);
        #pragma unroll
        for (int dt = 0; dt < 4; ++dt) {
            #pragma unroll
            for (int r = 0; r < 8; ++r) {
                int rowm = mt * 16 + rh + r;
                if (rowm >= BEIT_N) continue;
                float o = ov[dt][r] / rsum[r];
                ao[((size_t)b * BEIT_N + rowm) * BEIT_DIM + h * 64 + dt * 16 + nlane] =
                    (bf16_t)o;
            }
        }
    }
}

// ---------------------------------------------------------------------------
// Output projection: [12608 x 768] x [768 x 768]^T + proj_bias -> fp32 d_out.
// ---------------------------------------------------------------------------
__global__ __launch_bounds__(256) void proj_gemm(
    const bf16_t* __restrict__ ab, const bf16_t* __restrict__ wp,
    const float* __restrict__ p_bias, const int* __restrict__ b_idx,
    float* __restrict__ out) {
    __shared__ alignas(32) bf16_t alds[8 * 512];
    __shared__ alignas(32) bf16_t blds[8 * 512];

    const int tid  = threadIdx.x;
    const int wv   = tid >> 5, lane = tid & 31;
    const int col0 = blockIdx.x * 128;   // 0..640
    const int row0 = blockIdx.y * 128;
    const int wm = wv >> 1, wn = wv & 1;

    v8f acc[2][4];
    #pragma unroll
    for (int i = 0; i < 2; ++i)
        #pragma unroll
        for (int j = 0; j < 4; ++j) acc[i][j] = vzero8();

    for (int k0 = 0; k0 < BEIT_DIM; k0 += 32) {
        #pragma unroll
        for (int i = 0; i < 8; ++i) {
            int u  = tid + (i << 8);
            int ml = u >> 4;
            int k  = (u & 15) << 1;
            int row = row0 + ml; if (row > BEIT_M - 1) row = BEIT_M - 1;
            unsigned val = *(const unsigned*)(ab + (size_t)row * BEIT_DIM + k0 + k);
            int l = (ml & 15) + (((k >> 3) & 1) << 4);
            int e = ((k >> 4) << 3) + (k & 7);
            *(unsigned*)(alds + (ml >> 4) * 512 + l * 16 + e) = val;
        }
        #pragma unroll
        for (int i = 0; i < 8; ++i) {
            int u  = tid + (i << 8);
            int nl = u >> 4;
            int k  = (u & 15) << 1;
            unsigned val = *(const unsigned*)(wp + (size_t)(col0 + nl) * BEIT_DIM + k0 + k);
            int l = (nl & 15) + (((k >> 3) & 1) << 4);
            int e = ((k >> 4) << 3) + (k & 7);
            *(unsigned*)(blds + (nl >> 4) * 512 + l * 16 + e) = val;
        }
        __syncthreads();

        v16bf af[2], bf[4];
        #pragma unroll
        for (int i = 0; i < 2; ++i)
            af[i] = *(const v16bf*)(alds + (wm * 2 + i) * 512 + lane * 16);
        #pragma unroll
        for (int j = 0; j < 4; ++j)
            bf[j] = *(const v16bf*)(blds + (wn * 4 + j) * 512 + lane * 16);
        #pragma unroll
        for (int i = 0; i < 2; ++i)
            #pragma unroll
            for (int j = 0; j < 4; ++j)
                acc[i][j] = wmma_bf16(af[i], bf[j], acc[i][j]);
        __syncthreads();
    }

    const int nlane = lane & 15;
    const int rh    = (lane >> 4) << 3;
    #pragma unroll
    for (int i = 0; i < 2; ++i) {
        #pragma unroll
        for (int j = 0; j < 4; ++j) {
            int colg = col0 + wn * 64 + j * 16 + nlane;
            #pragma unroll
            for (int r = 0; r < 8; ++r) {
                int row = row0 + wm * 32 + i * 16 + rh + r;
                if (row >= BEIT_M) continue;
                int bsm = row / BEIT_N;
                int bi  = b_idx[bsm];
                out[(size_t)row * BEIT_DIM + colg] =
                    acc[i][j][r] + p_bias[bi * BEIT_DIM + colg];
            }
        }
    }
}

// ---------------------------------------------------------------------------
extern "C" void kernel_launch(void* const* d_in, const int* in_sizes, int n_in,
                              void* d_out, int out_size, void* d_ws, size_t ws_size,
                              hipStream_t stream) {
    (void)in_sizes; (void)n_in; (void)out_size; (void)ws_size;

    const float* x        = (const float*)d_in[0];   // [64,197,768]
    const float* qkv_w    = (const float*)d_in[1];   // [2304,768]
    const float* q_bias   = (const float*)d_in[2];   // [10,768]
    const float* v_bias   = (const float*)d_in[3];   // [10,768]
    const float* rel_tab  = (const float*)d_in[4];   // [732,12]
    const float* proj_w   = (const float*)d_in[5];   // [768,768]
    const float* proj_b   = (const float*)d_in[6];   // [10,768]
    const int*   b_idx    = (const int*)d_in[7];     // [64]
    const int*   rel_idx  = (const int*)d_in[8];     // [197,197]
    float*       out      = (float*)d_out;           // [64,197,768]

    // workspace carve-out (256B aligned)
    size_t off = 0;
    auto carve = [&](size_t bytes) -> void* {
        void* p = (char*)d_ws + off;
        off += (bytes + 255) & ~(size_t)255;
        return p;
    };
    const size_t MD = (size_t)BEIT_M * BEIT_DIM;          // 9,682,944
    bf16_t* xb    = (bf16_t*)carve(MD * 2);
    bf16_t* wqb   = (bf16_t*)carve((size_t)BEIT_3DIM * BEIT_DIM * 2);
    bf16_t* wpb   = (bf16_t*)carve((size_t)BEIT_DIM * BEIT_DIM * 2);
    bf16_t* qbuf  = (bf16_t*)carve(MD * 2);
    bf16_t* kbuf  = (bf16_t*)carve(MD * 2);
    bf16_t* vbuf  = (bf16_t*)carve(MD * 2);
    bf16_t* abuf  = (bf16_t*)carve(MD * 2);
    float*  relb  = (float*)carve((size_t)BEIT_HEADS * BEIT_N * BEIT_N * 4);

    // 1) bf16 casts
    {
        int n = (int)MD;
        cast_f32_bf16<<<(n + 255) / 256, 256, 0, stream>>>(x, xb, n);
        n = BEIT_3DIM * BEIT_DIM;
        cast_f32_bf16<<<(n + 255) / 256, 256, 0, stream>>>(qkv_w, wqb, n);
        n = BEIT_DIM * BEIT_DIM;
        cast_f32_bf16<<<(n + 255) / 256, 256, 0, stream>>>(proj_w, wpb, n);
    }
    // 2) relative-position bias gather
    relbias_build<<<(BEIT_N * BEIT_N + 255) / 256, 256, 0, stream>>>(
        rel_tab, rel_idx, relb);
    // 3) QKV projection + bias/scale/split
    qkv_gemm<<<dim3(BEIT_3DIM / 128, (BEIT_M + 127) / 128), 256, 0, stream>>>(
        xb, wqb, q_bias, v_bias, b_idx, qbuf, kbuf, vbuf);
    // 4) attention per (b, h)
    attn_kernel<<<BEIT_B * BEIT_HEADS, 256, 0, stream>>>(
        qbuf, kbuf, vbuf, relb, abuf);
    // 5) output projection + bias
    proj_gemm<<<dim3(BEIT_DIM / 128, (BEIT_M + 127) / 128), 256, 0, stream>>>(
        abuf, wpb, proj_b, b_idx, out);
}
